// GuidedMultiHeadAttention_17334488006965
// MI455X (gfx1250) — matrix operand order repaired and checked by reading, not compiled, not run on hardware
//
#include <hip/hip_runtime.h>

#define LSEQ 2048
#define EDIM 2048
#define NHEAD 16
#define DHEAD 128
#define THREE_E (3 * EDIM)

typedef __bf16 bf16_t;
typedef __attribute__((ext_vector_type(16))) __bf16 v16bf;
typedef __attribute__((ext_vector_type(8)))  __bf16 v8bf;
typedef __attribute__((ext_vector_type(8)))  float  v8f;

// ---------------------------------------------------------------------------
// Fragment loaders for NT GEMM (A and B both row-major, contiguous in K).
// A 16x32 bf16 layout: lanes 0-15  -> M=lane,    K = {k..k+7, k+16..k+23}
//                      lanes 16-31 -> M=lane-16, K = {k+8..k+15, k+24..k+31}
// B 32x16 bf16 layout: lanes 0-15  -> N=lane,    K = {k..k+7, k+16..k+23}
//                      lanes 16-31 -> N=lane-16, K = {k+8..k+15, k+24..k+31}
// ---------------------------------------------------------------------------
__device__ __forceinline__ v16bf load_fragA(const bf16_t* __restrict__ base,
                                            int row, int k, int ld, int lane) {
  const bf16_t* p = base + (size_t)(row + (lane & 15)) * ld + k + ((lane & 16) ? 8 : 0);
  union { v16bf v; v8bf h[2]; } f;
  f.h[0] = *(const v8bf*)(p);
  f.h[1] = *(const v8bf*)(p + 16);
  return f.v;
}

__device__ __forceinline__ v16bf load_fragB(const bf16_t* __restrict__ base,
                                            int col, int k, int ld, int lane) {
  const bf16_t* p = base + (size_t)(col + (lane & 15)) * ld + k + ((lane & 16) ? 8 : 0);
  union { v16bf v; v8bf h[2]; } f;
  f.h[0] = *(const v8bf*)(p);
  f.h[1] = *(const v8bf*)(p + 16);
  return f.v;
}

struct Acc { v8f c[4][4]; };

__device__ __forceinline__ void acc_zero(Acc& acc) {
#pragma unroll
  for (int i = 0; i < 4; ++i)
#pragma unroll
    for (int j = 0; j < 4; ++j)
      acc.c[i][j] = v8f{};
}

// One wave computes a 64x64 block of C = A * B^T (bf16 in, f32 accum).
__device__ __forceinline__ void gemm64x64_nt(const bf16_t* __restrict__ A,
                                             const bf16_t* __restrict__ B,
                                             int row0, int col0, int kdim,
                                             int lda, int ldb, int lane, Acc& acc) {
#pragma unroll 1
  for (int k = 0; k < kdim; k += 32) {
    v16bf a[4], b[4];
#pragma unroll
    for (int t = 0; t < 4; ++t) a[t] = load_fragA(A, row0 + 16 * t, k, lda, lane);
#pragma unroll
    for (int t = 0; t < 4; ++t) b[t] = load_fragB(B, col0 + 16 * t, k, ldb, lane);
#pragma unroll
    for (int i = 0; i < 4; ++i)
#pragma unroll
      for (int j = 0; j < 4; ++j)
        acc.c[i][j] = __builtin_amdgcn_wmma_f32_16x16x32_bf16(
            false, a[i], false, b[j], (short)0, acc.c[i][j], false, false);
  }
}

// C element mapping: VGPR v of lane l -> row = row0+16*i + v + 8*(l>=16),
//                                        col = col0+16*j + (l&15)

// ---------------------------------------------------------------------------
// fp32 -> bf16 conversion
// ---------------------------------------------------------------------------
__global__ void k_f32_to_bf16(const float* __restrict__ in, bf16_t* __restrict__ out, int n) {
  int i = blockIdx.x * 256 + threadIdx.x;
  if (i < n) out[i] = (bf16_t)in[i];
}

// ---------------------------------------------------------------------------
// QKV projection: qkv = x @ Win^T + bin, scatter into Q[H][L][D] (pre-scaled),
// K[H][L][D], Vt[H][D][L] (transposed for the attn*V NT GEMM).
// ---------------------------------------------------------------------------
__global__ void __launch_bounds__(128) k_qkv(const bf16_t* __restrict__ xb,
                                             const bf16_t* __restrict__ Winb,
                                             const float* __restrict__ bin,
                                             bf16_t* __restrict__ Q,
                                             bf16_t* __restrict__ Kd,
                                             bf16_t* __restrict__ Vt) {
  const int lane = threadIdx.x & 31;
  const int wave = threadIdx.x >> 5;
  const int row0 = blockIdx.y * 128 + (wave >> 1) * 64;
  const int col0 = blockIdx.x * 128 + (wave & 1) * 64;
  Acc acc; acc_zero(acc);
  gemm64x64_nt(xb, Winb, row0, col0, EDIM, EDIM, EDIM, lane, acc);

  const float scale = 0.08838834764831845f;  // 1/sqrt(128)
#pragma unroll
  for (int i = 0; i < 4; ++i) {
    const int rbase = row0 + 16 * i + ((lane >> 4) << 3);
#pragma unroll
    for (int j = 0; j < 4; ++j) {
      const int ncol = col0 + 16 * j + (lane & 15);
      const float bc = bin[ncol];
#pragma unroll
      for (int v = 0; v < 8; ++v) {
        const float val = acc.c[i][j][v] + bc;
        const int r = rbase + v;
        if (ncol < EDIM) {
          const int h = ncol >> 7, d = ncol & 127;
          Q[((size_t)h * LSEQ + r) * DHEAD + d] = (bf16_t)(val * scale);
        } else if (ncol < 2 * EDIM) {
          const int c = ncol - EDIM;
          const int h = c >> 7, d = c & 127;
          Kd[((size_t)h * LSEQ + r) * DHEAD + d] = (bf16_t)val;
        } else {
          const int c = ncol - 2 * EDIM;
          const int h = c >> 7, d = c & 127;
          Vt[((size_t)h * DHEAD + d) * LSEQ + r] = (bf16_t)val;
        }
      }
    }
  }
}

// ---------------------------------------------------------------------------
// Scores: S[z] = Q[head0+z] @ K[head0+z]^T + bias   (f32, [L][L] per z-slice)
// z = blockIdx.z; S slice stride passed so the same kernel serves both the
// batched (all heads, one launch) and looped (reused buffer) schedules.
// ---------------------------------------------------------------------------
__global__ void __launch_bounds__(128) k_scores(const bf16_t* __restrict__ Q,
                                                const bf16_t* __restrict__ Kd,
                                                const float* __restrict__ biasM,
                                                float* __restrict__ S,
                                                int head0, unsigned long long sStride) {
  const int h = head0 + blockIdx.z;
  const bf16_t* Qh = Q  + (size_t)h * LSEQ * DHEAD;
  const bf16_t* Kh = Kd + (size_t)h * LSEQ * DHEAD;
  float* Sz = S + (size_t)blockIdx.z * sStride;

  const int lane = threadIdx.x & 31;
  const int wave = threadIdx.x >> 5;
  const int row0 = blockIdx.y * 128 + (wave >> 1) * 64;
  const int col0 = blockIdx.x * 128 + (wave & 1) * 64;
  Acc acc; acc_zero(acc);
  gemm64x64_nt(Qh, Kh, row0, col0, DHEAD, DHEAD, DHEAD, lane, acc);

#pragma unroll
  for (int i = 0; i < 4; ++i) {
    const int rbase = row0 + 16 * i + ((lane >> 4) << 3);
#pragma unroll
    for (int j = 0; j < 4; ++j) {
      const int ncol = col0 + 16 * j + (lane & 15);
#pragma unroll
      for (int v = 0; v < 8; ++v) {
        const size_t idx = (size_t)(rbase + v) * LSEQ + ncol;
        Sz[idx] = acc.c[i][j][v] + biasM[idx];
      }
    }
  }
}

// ---------------------------------------------------------------------------
// Row softmax: S f32 -> P bf16. grid = (L, Z); one 256-thread block per row.
// ---------------------------------------------------------------------------
__global__ void __launch_bounds__(256) k_softmax(const float* __restrict__ S,
                                                 bf16_t* __restrict__ P,
                                                 unsigned long long sStride,
                                                 unsigned long long pStride) {
  __shared__ float sm[8];
  const int row = blockIdx.x;
  const int tid = threadIdx.x;
  const float* s = S + (size_t)blockIdx.y * sStride + (size_t)row * LSEQ;

  float v[8];
  float m = -3.4e38f;
#pragma unroll
  for (int t = 0; t < 8; ++t) { v[t] = s[tid + 256 * t]; m = fmaxf(m, v[t]); }
#pragma unroll
  for (int o = 16; o > 0; o >>= 1) m = fmaxf(m, __shfl_xor(m, o, 32));
  if ((tid & 31) == 0) sm[tid >> 5] = m;
  __syncthreads();
  m = sm[0];
#pragma unroll
  for (int i = 1; i < 8; ++i) m = fmaxf(m, sm[i]);
  __syncthreads();

  float sum = 0.0f;
#pragma unroll
  for (int t = 0; t < 8; ++t) { v[t] = __expf(v[t] - m); sum += v[t]; }
#pragma unroll
  for (int o = 16; o > 0; o >>= 1) sum += __shfl_xor(sum, o, 32);
  if ((tid & 31) == 0) sm[tid >> 5] = sum;
  __syncthreads();
  sum = 0.0f;
#pragma unroll
  for (int i = 0; i < 8; ++i) sum += sm[i];
  const float inv = 1.0f / sum;

  bf16_t* p = P + (size_t)blockIdx.y * pStride + (size_t)row * LSEQ;
#pragma unroll
  for (int t = 0; t < 8; ++t) p[tid + 256 * t] = (bf16_t)(v[t] * inv);
}

// ---------------------------------------------------------------------------
// ctx[head0+z] = P[z] @ Vt[head0+z]^T : C[i,d] = sum_j P[i,j] * Vt[d,j].
// N = DHEAD = 128. Writes bf16 into ctx[L][E] at column offset head*DHEAD.
// ---------------------------------------------------------------------------
__global__ void __launch_bounds__(128) k_ctx(const bf16_t* __restrict__ P,
                                             const bf16_t* __restrict__ Vt,
                                             bf16_t* __restrict__ ctx,
                                             int head0, unsigned long long pStride) {
  const int h = head0 + blockIdx.z;
  const bf16_t* Pz  = P  + (size_t)blockIdx.z * pStride;
  const bf16_t* Vth = Vt + (size_t)h * DHEAD * LSEQ;

  const int lane = threadIdx.x & 31;
  const int wave = threadIdx.x >> 5;
  const int row0 = blockIdx.y * 128 + (wave >> 1) * 64;
  const int col0 = (wave & 1) * 64;  // N = 128 -> single block in N
  Acc acc; acc_zero(acc);
  gemm64x64_nt(Pz, Vth, row0, col0, LSEQ, LSEQ, LSEQ, lane, acc);

#pragma unroll
  for (int i = 0; i < 4; ++i) {
    const int rbase = row0 + 16 * i + ((lane >> 4) << 3);
#pragma unroll
    for (int j = 0; j < 4; ++j) {
      const int d = col0 + 16 * j + (lane & 15);
#pragma unroll
      for (int v = 0; v < 8; ++v)
        ctx[(size_t)(rbase + v) * EDIM + h * DHEAD + d] = (bf16_t)acc.c[i][j][v];
    }
  }
}

// ---------------------------------------------------------------------------
// out = ctx @ Wout^T + bout  (f32 out, [L][E])
// ---------------------------------------------------------------------------
__global__ void __launch_bounds__(128) k_out(const bf16_t* __restrict__ ctx,
                                             const bf16_t* __restrict__ Woutb,
                                             const float* __restrict__ bout,
                                             float* __restrict__ out) {
  const int lane = threadIdx.x & 31;
  const int wave = threadIdx.x >> 5;
  const int row0 = blockIdx.y * 128 + (wave >> 1) * 64;
  const int col0 = blockIdx.x * 128 + (wave & 1) * 64;
  Acc acc; acc_zero(acc);
  gemm64x64_nt(ctx, Woutb, row0, col0, EDIM, EDIM, EDIM, lane, acc);

#pragma unroll
  for (int i = 0; i < 4; ++i) {
    const int rbase = row0 + 16 * i + ((lane >> 4) << 3);
#pragma unroll
    for (int j = 0; j < 4; ++j) {
      const int ncol = col0 + 16 * j + (lane & 15);
      const float bc = bout[ncol];
#pragma unroll
      for (int v = 0; v < 8; ++v)
        out[(size_t)(rbase + v) * EDIM + ncol] = acc.c[i][j][v] + bc;
    }
  }
}

// ---------------------------------------------------------------------------
extern "C" void kernel_launch(void* const* d_in, const int* in_sizes, int n_in,
                              void* d_out, int out_size, void* d_ws, size_t ws_size,
                              hipStream_t stream) {
  const float* x      = (const float*)d_in[0];
  const float* biasM  = (const float*)d_in[1];
  const float* Win    = (const float*)d_in[2];
  const float* bin    = (const float*)d_in[3];
  const float* Wout   = (const float*)d_in[4];
  const float* bout   = (const float*)d_in[5];
  float* out = (float*)d_out;

  char* ws = (char*)d_ws;
  size_t off = 0;
  auto carve = [&](size_t bytes) {
    void* p = ws + off;
    off += (bytes + 255) & ~(size_t)255;
    return p;
  };

  bf16_t* xb    = (bf16_t*)carve((size_t)LSEQ * EDIM * 2);      // 8 MB
  bf16_t* Winb  = (bf16_t*)carve((size_t)THREE_E * EDIM * 2);   // 24 MB
  bf16_t* Woutb = (bf16_t*)carve((size_t)EDIM * EDIM * 2);      // 8 MB
  bf16_t* Q     = (bf16_t*)carve((size_t)LSEQ * EDIM * 2);      // 8 MB  [H][L][D]
  bf16_t* Kd    = (bf16_t*)carve((size_t)LSEQ * EDIM * 2);      // 8 MB  [H][L][D]
  bf16_t* Vt    = (bf16_t*)carve((size_t)LSEQ * EDIM * 2);      // 8 MB  [H][D][L]
  bf16_t* ctx   = (bf16_t*)carve((size_t)LSEQ * EDIM * 2);      // 8 MB  [L][E]

  const size_t fixed = off;
  const size_t sSliceB = (size_t)LSEQ * LSEQ * sizeof(float);   // 16 MB
  const size_t pSliceB = (size_t)LSEQ * LSEQ * sizeof(bf16_t);  //  8 MB
  // Batched schedule (one launch per stage, heads on blockIdx.z) needs
  // full [H][L][L] score + prob buffers; otherwise loop heads, reusing one slice.
  const bool batched =
      ws_size >= fixed + (size_t)NHEAD * (sSliceB + pSliceB) + 512;
  const int Z = batched ? NHEAD : 1;

  float*  S = (float*) carve(batched ? (size_t)NHEAD * sSliceB : sSliceB);
  bf16_t* P = (bf16_t*)carve(batched ? (size_t)NHEAD * pSliceB : pSliceB);

  const unsigned long long sStride = batched ? (unsigned long long)LSEQ * LSEQ : 0ull;
  const unsigned long long pStride = batched ? (unsigned long long)LSEQ * LSEQ : 0ull;

  // fp32 -> bf16 conversions
  {
    int n = LSEQ * EDIM;
    k_f32_to_bf16<<<(n + 255) / 256, 256, 0, stream>>>(x, xb, n);
  }
  {
    int n = THREE_E * EDIM;
    k_f32_to_bf16<<<(n + 255) / 256, 256, 0, stream>>>(Win, Winb, n);
  }
  {
    int n = EDIM * EDIM;
    k_f32_to_bf16<<<(n + 255) / 256, 256, 0, stream>>>(Wout, Woutb, n);
  }

  // QKV projection: [L,E] x [3E,E]^T -> Q/K/Vt
  k_qkv<<<dim3(THREE_E / 128, LSEQ / 128), 128, 0, stream>>>(xb, Winb, bin, Q, Kd, Vt);

  // Attention: batched = 3 launches total; looped = 3 launches per head.
  const int iters = batched ? 1 : NHEAD;
  for (int h0 = 0; h0 < iters; ++h0) {
    k_scores<<<dim3(LSEQ / 128, LSEQ / 128, Z), 128, 0, stream>>>(
        Q, Kd, biasM, S, h0, sStride);
    k_softmax<<<dim3(LSEQ, Z), 256, 0, stream>>>(S, P, sStride, pStride);
    k_ctx<<<dim3(1, LSEQ / 128, Z), 128, 0, stream>>>(P, Vt, ctx, h0, pStride);
  }

  // Output projection
  k_out<<<dim3(EDIM / 128, LSEQ / 128), 128, 0, stream>>>(ctx, Woutb, bout, out);
}
